// BSAM_60129542251
// MI455X (gfx1250) — matrix operand order repaired and checked by reading, not compiled, not run on hardware
//
#include <hip/hip_runtime.h>
#include <hip/hip_bf16.h>

typedef __attribute__((ext_vector_type(16))) __bf16 v16bf;
typedef __attribute__((ext_vector_type(8)))  __bf16 v8bf;
typedef __attribute__((ext_vector_type(4)))  __bf16 v4bf;
typedef __attribute__((ext_vector_type(8)))  float  v8f;
typedef __attribute__((ext_vector_type(2)))  int    i32x2;
typedef __attribute__((ext_vector_type(4)))  int    i32x4;

#define N_PIX 4096
#define HW    64
#define CIN   64
#define DQK   32
#define DV    64
#define NB    4

// LDS row strides (elements). 80B rows: 16B-aligned b128 accesses, banks spread.
#define KSTRIDE 40
#define VSTRIDE 40

// ---- gfx1250 async global->LDS path, guarded so compile never breaks -------
#if defined(__HIP_DEVICE_COMPILE__) && \
    __has_builtin(__builtin_amdgcn_global_load_async_to_lds_b64) && \
    __has_builtin(__builtin_amdgcn_global_load_async_to_lds_b128)
#define USE_ASYNC 1
#else
#define USE_ASYNC 0
#endif

// typed, address-space-qualified pointer casts (signature from hipcc diag:
// param0 = v2i/v4i in AS1 "__device__", param1 = LDS side, then imm offset, imm cpol)
#define GPTR2(p) ((__attribute__((address_space(1))) i32x2*)(p))
#define LPTR2(p) ((__attribute__((address_space(3))) i32x2*)(p))
#define GPTR4(p) ((__attribute__((address_space(1))) i32x4*)(p))
#define LPTR4(p) ((__attribute__((address_space(3))) i32x4*)(p))

#if USE_ASYNC
#if __has_builtin(__builtin_amdgcn_s_wait_asynccnt)
#define WAIT_ASYNC(n) __builtin_amdgcn_s_wait_asynccnt(n)
#else
#define WAIT_ASYNC(n) asm volatile("s_wait_asynccnt %0" :: "i"(n) : "memory")
#endif
#else
#define WAIT_ASYNC(n) ((void)0)
#endif

// ---------------------------------------------------------------------------
// Direct 3x3 conv (stride 1, pad 1). One block = 16x16 spatial tile of one
// batch image; all COUT channels accumulate in registers; 18x18 halo tile per
// input channel staged in LDS; weights are block-uniform (scalar loads).
// Output bf16 in the layout the attention kernel wants.
// ---------------------------------------------------------------------------
template<int COUT, int TRANSPOSED>
__global__ __launch_bounds__(256) void conv3x3_bf16(
        const float* __restrict__ in, const float* __restrict__ w,
        const float* __restrict__ bias, __bf16* __restrict__ out)
{
    __shared__ float smem[18 * 18];
    const int tid  = threadIdx.x;
    const int b    = blockIdx.x >> 4;
    const int tile = blockIdx.x & 15;
    const int ty0  = (tile >> 2) * 16;
    const int tx0  = (tile & 3) * 16;
    const int px   = tid & 15, py = tid >> 4;

    float acc[COUT];
    #pragma unroll
    for (int oc = 0; oc < COUT; ++oc) acc[oc] = bias[oc];

    for (int ic = 0; ic < CIN; ++ic) {
        const float* inp = in + ((size_t)(b * CIN + ic)) * N_PIX;
        for (int i = tid; i < 18 * 18; i += 256) {
            int r = i / 18, c = i % 18;
            int gy = ty0 + r - 1, gx = tx0 + c - 1;
            float v = 0.f;
            if (gy >= 0 && gy < HW && gx >= 0 && gx < HW) v = inp[gy * HW + gx];
            smem[i] = v;
        }
        __syncthreads();
        #pragma unroll
        for (int ky = 0; ky < 3; ++ky)
            #pragma unroll
            for (int kx = 0; kx < 3; ++kx) {
                float v = smem[(py + ky) * 18 + (px + kx)];
                const float* wp = w + (size_t)ic * 9 + ky * 3 + kx;  // + oc*CIN*9
                #pragma unroll
                for (int oc = 0; oc < COUT; ++oc)
                    acc[oc] = fmaf(v, wp[(size_t)oc * CIN * 9], acc[oc]);
            }
        __syncthreads();
    }

    const int pix = (ty0 + py) * HW + (tx0 + px);
    #pragma unroll
    for (int oc = 0; oc < COUT; ++oc) {
        size_t idx = TRANSPOSED ? ((size_t)b * N_PIX + pix) * COUT + oc
                                : ((size_t)(b * COUT + oc)) * N_PIX + pix;
        out[idx] = (__bf16)acc[oc];
    }
}

// ---------------------------------------------------------------------------
// Cooperative stage of one 32-key block of K (32x32 bf16, contiguous 2KB) and
// V^T (64 chans x 32 keys, row stride N_PIX) into padded LDS buffers.
// 256 threads: 8B of K + 16B of V each. Async (ASYNCcnt) when available.
// ---------------------------------------------------------------------------
__device__ __forceinline__ void stage_kv(const __bf16* __restrict__ kg,
                                         const __bf16* __restrict__ vg,
                                         __bf16* kl, __bf16* vl, int t)
{
    {   // K block: fully contiguous in global
        int row = t >> 3, part = t & 7;
        const __bf16* src = kg + t * 4;
        __bf16* dst = kl + row * KSTRIDE + part * 4;
#if USE_ASYNC
        __builtin_amdgcn_global_load_async_to_lds_b64(GPTR2(src), LPTR2(dst), 0, 0);
#else
        *(v4bf*)dst = *(const v4bf*)src;
#endif
    }
    {   // V block: 64 rows of 64B, row stride N_PIX elements
        int chan = t >> 2, part = t & 3;
        const __bf16* src = vg + (size_t)chan * N_PIX + part * 8;
        __bf16* dst = vl + chan * VSTRIDE + part * 8;
#if USE_ASYNC
        __builtin_amdgcn_global_load_async_to_lds_b128(GPTR4(src), LPTR4(dst), 0, 0);
#else
        *(v8bf*)dst = *(const v8bf*)src;
#endif
    }
}

// ---------------------------------------------------------------------------
// Fused flash attention. 8 waves/WG, one 16-query tile per wave, all waves
// share double-buffered K/V LDS staging (async copies overlap compute).
//   S^T tile = K(16k x 32f) x Q^T(32f x 16q)   v_wmma_f32_16x16x32_bf16
//   online softmax in f32, P^T via one lane L<->L^16 exchange
//   E^T tile += V^T(16c x 32k) x P^T(32k x 16q)
// ---------------------------------------------------------------------------
__global__ __launch_bounds__(256) void flash_attn_bf16(
        const __bf16* __restrict__ Q, const __bf16* __restrict__ K,
        const __bf16* __restrict__ Vt, const float* __restrict__ res,
        float* __restrict__ out)
{
    __shared__ __bf16 kbuf[2][32 * KSTRIDE];
    __shared__ __bf16 vbuf[2][DV * VSTRIDE];

    const int tid   = threadIdx.x;
    const int lane  = tid & 31;
    const int hl    = lane >> 4;
    const int l15   = lane & 15;
    const int wv    = tid >> 5;                       // wave in workgroup
    const int b     = blockIdx.x >> 5;                // 32 blocks per batch
    const int qBase = (((blockIdx.x & 31) << 3) + wv) << 4;

    // Q as B-operand (32f x 16q); loop-invariant, kept in registers.
    const __bf16* qp = Q + ((size_t)b * N_PIX + qBase + l15) * DQK + hl * 8;
    v8bf qlo = *(const v8bf*)qp;
    v8bf qhi = *(const v8bf*)(qp + 16);
    v16bf qb;
    #pragma unroll
    for (int i = 0; i < 8; ++i) { qb[i] = qlo[i]; qb[8 + i] = qhi[i]; }

    const __bf16* kb0 = K  + (size_t)b * N_PIX * DQK;
    const __bf16* vb0 = Vt + (size_t)b * DV * N_PIX;

    v8f acc[4] = {};
    v8f czero  = {};
    float mrun = -3.0e38f, lrun = 0.f;

    stage_kv(kb0, vb0, kbuf[0], vbuf[0], tid);        // prologue: block 0

    for (int kb = 0; kb < N_PIX; kb += 32) {
        const int buf = (kb >> 5) & 1;
        const bool more = (kb + 32) < N_PIX;
        if (more)                                      // overlap next copy
            stage_kv(kb0 + (size_t)(kb + 32) * DQK, vb0 + kb + 32,
                     kbuf[buf ^ 1], vbuf[buf ^ 1], tid);
#if USE_ASYNC
        if (more) WAIT_ASYNC(2); else WAIT_ASYNC(0);   // drain current block
#endif
        __syncthreads();                               // current block visible

        // ---- K A-operands from LDS ----
        const __bf16* krow = &kbuf[buf][(size_t)l15 * KSTRIDE + hl * 8];
        v8bf k0lo = *(const v8bf*)krow;
        v8bf k0hi = *(const v8bf*)(krow + 16);
        v8bf k1lo = *(const v8bf*)(krow + 16 * KSTRIDE);
        v8bf k1hi = *(const v8bf*)(krow + 16 * KSTRIDE + 16);
        v16bf ka0, ka1;
        #pragma unroll
        for (int i = 0; i < 8; ++i) {
            ka0[i] = k0lo[i]; ka0[8 + i] = k0hi[i];
            ka1[i] = k1lo[i]; ka1[8 + i] = k1hi[i];
        }

        // ---- V^T A-operands from LDS (issued early; consumed after softmax) --
        v16bf va[4];
        #pragma unroll
        for (int t = 0; t < 4; ++t) {
            const __bf16* vrow = &vbuf[buf][(size_t)(t * 16 + l15) * VSTRIDE + hl * 16];
            v8bf vlo = *(const v8bf*)vrow;
            v8bf vhi = *(const v8bf*)(vrow + 8);
            #pragma unroll
            for (int i = 0; i < 8; ++i) { va[t][i] = vlo[i]; va[t][8 + i] = vhi[i]; }
        }

        v8f s0 = __builtin_amdgcn_wmma_f32_16x16x32_bf16(
                     false, ka0, false, qb, (short)0, czero, false, false);
        v8f s1 = __builtin_amdgcn_wmma_f32_16x16x32_bf16(
                     false, ka1, false, qb, (short)0, czero, false, false);

        // ---- online softmax over this 32-key block (per-query = per-lane) ----
        float bm = -3.0e38f;
        #pragma unroll
        for (int r = 0; r < 8; ++r) { bm = fmaxf(bm, s0[r]); bm = fmaxf(bm, s1[r]); }
        bm = fmaxf(bm, __shfl_xor(bm, 16, 32));
        float mnew  = fmaxf(mrun, bm);
        float alpha = __expf(mrun - mnew);
        float p0v[8], p1v[8], psum = 0.f;
        #pragma unroll
        for (int r = 0; r < 8; ++r) {
            p0v[r] = __expf(s0[r] - mnew);
            p1v[r] = __expf(s1[r] - mnew);
            psum  += p0v[r] + p1v[r];
        }
        psum += __shfl_xor(psum, 16, 32);
        lrun  = lrun * alpha + psum;
        mrun  = mnew;

        // ---- P^T -> B-operand (32k x 16q): one lane L <-> L^16 exchange ----
        v16bf pb;
        #pragma unroll
        for (int r = 0; r < 8; ++r) {
            float o0 = __shfl_xor(p0v[r], 16, 32);     // partner keys 8..15
            float o1 = __shfl_xor(p1v[r], 16, 32);     // partner keys 16..23
            float lo = hl ? o1 : p0v[r];
            float hi = hl ? p1v[r] : o0;
            pb[r]     = (__bf16)lo;
            pb[8 + r] = (__bf16)hi;
        }

        // rescale running accumulators
        #pragma unroll
        for (int t = 0; t < 4; ++t)
            #pragma unroll
            for (int r = 0; r < 8; ++r) acc[t][r] *= alpha;

        // ---- E^T += V^T x P^T ----
        #pragma unroll
        for (int t = 0; t < 4; ++t)
            acc[t] = __builtin_amdgcn_wmma_f32_16x16x32_bf16(
                         false, va[t], false, pb, (short)0, acc[t], false, false);

        __syncthreads();           // all waves done with buf before overwrite
    }

    // ---- epilogue: normalize, add residual, store f32 [b][c][n] ----
    const float linv = 1.0f / lrun;
    #pragma unroll
    for (int t = 0; t < 4; ++t)
        #pragma unroll
        for (int r = 0; r < 8; ++r) {
            int c = t * 16 + hl * 8 + r;
            size_t idx = ((size_t)(b * DV + c)) * N_PIX + qBase + l15;
            out[idx] = acc[t][r] * linv + res[idx];
        }
}

// ---------------------------------------------------------------------------
extern "C" void kernel_launch(void* const* d_in, const int* in_sizes, int n_in,
                              void* d_out, int out_size, void* d_ws, size_t ws_size,
                              hipStream_t stream)
{
    const float* A1_B = (const float*)d_in[0];
    const float* A1_C = (const float*)d_in[1];
    const float* w1   = (const float*)d_in[2];
    const float* b1   = (const float*)d_in[3];
    const float* w2   = (const float*)d_in[4];
    const float* b2   = (const float*)d_in[5];
    const float* w3   = (const float*)d_in[6];
    const float* b3   = (const float*)d_in[7];
    float* out = (float*)d_out;

    char* ws = (char*)d_ws;
    __bf16* q_ws = (__bf16*)(ws);                    // [4][4096][32] bf16 : 1 MiB
    __bf16* k_ws = (__bf16*)(ws + (1u << 20));       // [4][4096][32] bf16 : 1 MiB
    __bf16* v_ws = (__bf16*)(ws + (2u << 20));       // [4][64][4096] bf16 : 2 MiB

    dim3 blk(256);
    conv3x3_bf16<32, 1><<<NB * 16, blk, 0, stream>>>(A1_B, w1, b1, q_ws);
    conv3x3_bf16<32, 1><<<NB * 16, blk, 0, stream>>>(A1_C, w2, b2, k_ws);
    conv3x3_bf16<64, 0><<<NB * 16, blk, 0, stream>>>(A1_C, w3, b3, v_ws);
    flash_attn_bf16<<<128, blk, 0, stream>>>(q_ws, k_ws, v_ws, A1_C, out);
}
